// TransformerBlock_58961311039674
// MI455X (gfx1250) — matrix operand order repaired and checked by reading, not compiled, not run on hardware
//
#include <hip/hip_runtime.h>
#include <hip/hip_bf16.h>
#include <math.h>

// ---------------------------------------------------------------------------
// CDNA5 (gfx1250) transformer block:
//   - all matmuls via v_wmma_f32_16x16x32_bf16
//   - tiles staged with global_load_async_to_lds_b128 (ASYNCcnt),
//     double-buffered LDS, one barrier per K-step (GEMM and attention)
// ---------------------------------------------------------------------------

typedef __attribute__((ext_vector_type(16))) __bf16 bf16x16;
typedef __attribute__((ext_vector_type(8)))  float  floatx8;

#define WMMA_BF16(A_, B_, C_) \
  __builtin_amdgcn_wmma_f32_16x16x32_bf16(false, (A_), false, (B_), (short)0, (C_), false, false)

static constexpr int Bb   = 4;
static constexpr int Ss   = 2048;
static constexpr int Dd   = 1024;
static constexpr int Hh   = 16;
static constexpr int DKk  = 64;
static constexpr int DFFf = 4096;
static constexpr int NTOK = Bb * Ss;

static constexpr int FLAG_GELU = 1;
static constexpr int FLAG_HEAD = 2;

__device__ __forceinline__ void async_copy_b128(const void* gsrc, void* lds_dst) {
  // GLOBAL_LOAD_ASYNC_TO_LDS_B128: per-lane 16B memory->LDS, tracked by ASYNCcnt.
  // Generic LDS pointer low 32 bits == LDS byte offset (ISA: addr truncated).
  uint32_t l = (uint32_t)(uintptr_t)lds_dst;
  uint64_t g = (uint64_t)(uintptr_t)gsrc;
  asm volatile("global_load_async_to_lds_b128 %0, %1, off" :: "v"(l), "v"(g) : "memory");
}
__device__ __forceinline__ void wait_async0() {
  asm volatile("s_wait_asynccnt 0" ::: "memory");
}

// ---------------------------------------------------------------------------
// fp32 -> bf16 conversion (weights)
// ---------------------------------------------------------------------------
__global__ void cvt_f32_bf16(const float* __restrict__ src, __bf16* __restrict__ dst, int n) {
  int i = blockIdx.x * blockDim.x + threadIdx.x;
  int stride = gridDim.x * blockDim.x;
  int n4 = n >> 2;
  for (int i4 = i; i4 < n4; i4 += stride) {
    float4 v = ((const float4*)src)[i4];
    dst[i4 * 4 + 0] = (__bf16)v.x;
    dst[i4 * 4 + 1] = (__bf16)v.y;
    dst[i4 * 4 + 2] = (__bf16)v.z;
    dst[i4 * 4 + 3] = (__bf16)v.w;
  }
}

// ---------------------------------------------------------------------------
// LayerNorm (row = token), fp32 in -> bf16 out.  D = 1024, block = 256.
// ---------------------------------------------------------------------------
__global__ void layernorm_bf16(const float* __restrict__ x, const float* __restrict__ g,
                               const float* __restrict__ beta, __bf16* __restrict__ out) {
  __shared__ float red[256];
  const int row = blockIdx.x;
  const int tid = threadIdx.x;
  float4 v = ((const float4*)(x + (size_t)row * Dd))[tid];
  float s = v.x + v.y + v.z + v.w;
  red[tid] = s; __syncthreads();
  for (int o = 128; o > 0; o >>= 1) { if (tid < o) red[tid] += red[tid + o]; __syncthreads(); }
  const float mean = red[0] * (1.0f / Dd);
  __syncthreads();
  float dx = v.x - mean, dy = v.y - mean, dz = v.z - mean, dw = v.w - mean;
  red[tid] = dx * dx + dy * dy + dz * dz + dw * dw;
  __syncthreads();
  for (int o = 128; o > 0; o >>= 1) { if (tid < o) red[tid] += red[tid + o]; __syncthreads(); }
  const float rstd = rsqrtf(red[0] * (1.0f / Dd) + 1e-5f);
  float4 g4 = ((const float4*)g)[tid];
  float4 b4 = ((const float4*)beta)[tid];
  __bf16* outr = out + (size_t)row * Dd + tid * 4;
  outr[0] = (__bf16)(dx * rstd * g4.x + b4.x);
  outr[1] = (__bf16)(dy * rstd * g4.y + b4.y);
  outr[2] = (__bf16)(dz * rstd * g4.z + b4.z);
  outr[3] = (__bf16)(dw * rstd * g4.w + b4.w);
}

// ---------------------------------------------------------------------------
// Tiled bf16 GEMM with WMMA:  C[M,N] = A[M,K] @ B[K,N] + bias (+epilogue).
// Block tile 128x128, 8 waves (4x2), each wave 32x64 = 2x4 WMMA tiles.
// K step 32, double-buffered LDS, async A staging.
// ---------------------------------------------------------------------------
__global__ void gemm_bf16_wmma(const __bf16* __restrict__ A, const __bf16* __restrict__ Bm,
                               const float* __restrict__ bias, const float* __restrict__ resid,
                               float* __restrict__ outF, __bf16* __restrict__ outBF,
                               int M, int N, int K, int flags) {
  // pitch 40 bf16 = 80 B: 16B-aligned segments, 20-dword bank stride (conflict-free rows)
  __shared__ __align__(16) __bf16 As[2][128][40];   // A tile, row-major
  __shared__ __align__(16) __bf16 Bt[2][128][40];   // B tile, transposed [n][k]

  const int t    = threadIdx.x;
  const int wave = t >> 5;
  const int lane = t & 31;
  const int wm   = wave >> 1;          // 0..3 -> 32-row slab
  const int wn   = wave & 1;           // 0..1 -> 64-col slab
  const int lr   = lane & 15;
  const int hi   = lane >> 4;
  const int lk   = hi * 16;
  const int bk   = hi * 16;

  const int n0 = blockIdx.x * 128;
  const int m0 = blockIdx.y * 128;

  floatx8 acc[2][4];
#pragma unroll
  for (int a = 0; a < 2; ++a)
#pragma unroll
    for (int b = 0; b < 4; ++b)
#pragma unroll
      for (int r = 0; r < 8; ++r) acc[a][b][r] = 0.f;

  // staging helpers ---------------------------------------------------------
  const int sAr = t >> 2;            // 0..63
  const int sAc = (t & 3) * 8;       // 0,8,16,24
  const int sBk = t >> 3;            // 0..31
  const int sBn = (t & 7) * 16;      // 0..112

  auto stage = [&](int kk, int buf) {
    // A: 128x32, two 16B async copies per thread
#pragma unroll
    for (int it = 0; it < 2; ++it) {
      int r = it * 64 + sAr;
      async_copy_b128(A + (size_t)(m0 + r) * K + kk + sAc, &As[buf][r][sAc]);
    }
    // B: 32x128 -> transposed Bt[n][k], 16 elements per thread
    const __bf16* src = Bm + (size_t)(kk + sBk) * N + n0 + sBn;
#pragma unroll
    for (int j = 0; j < 16; ++j) Bt[buf][sBn + j][sBk] = src[j];
  };

  const int KT = K >> 5;
  stage(0, 0);
  wait_async0();
  __syncthreads();

  for (int kt = 0; kt < KT; ++kt) {
    const int cur = kt & 1;
    if (kt + 1 < KT) stage((kt + 1) << 5, cur ^ 1);   // overlap next tile with math

    bf16x16 af[2], bfr[4];
#pragma unroll
    for (int ms = 0; ms < 2; ++ms) {
      int row = wm * 32 + ms * 16 + lr;
#pragma unroll
      for (int j = 0; j < 8; ++j) { af[ms][j] = As[cur][row][lk + j]; af[ms][8 + j] = As[cur][row][lk + 8 + j]; }
    }
#pragma unroll
    for (int ns = 0; ns < 4; ++ns) {
      int col = wn * 64 + ns * 16 + lr;
#pragma unroll
      for (int e = 0; e < 16; ++e) bfr[ns][e] = Bt[cur][col][bk + e];
    }
#pragma unroll
    for (int ms = 0; ms < 2; ++ms)
#pragma unroll
      for (int ns = 0; ns < 4; ++ns)
        acc[ms][ns] = WMMA_BF16(af[ms], bfr[ns], acc[ms][ns]);

    wait_async0();
    __syncthreads();
  }

  // ---- epilogue ----
#pragma unroll
  for (int ms = 0; ms < 2; ++ms) {
#pragma unroll
    for (int ns = 0; ns < 4; ++ns) {
#pragma unroll
      for (int r = 0; r < 8; ++r) {
        int grow = m0 + wm * 32 + ms * 16 + hi * 8 + r;
        int gcol = n0 + wn * 64 + ns * 16 + lr;
        float v = acc[ms][ns][r] + bias[gcol];
        if (flags & FLAG_GELU) v = 0.5f * v * (1.0f + erff(v * 0.70710678118654752f));
        if (resid) v += resid[(size_t)grow * N + gcol];
        if (outF)  outF[(size_t)grow * N + gcol] = v;
        if (outBF) {
          if (flags & FLAG_HEAD) {   // store into [B,H,S,DK]
            int b  = grow >> 11, s = grow & 2047;
            int h  = gcol >> 6,  dk = gcol & 63;
            outBF[(((size_t)(b * Hh + h)) * Ss + s) * DKk + dk] = (__bf16)v;
          } else {
            outBF[(size_t)grow * N + gcol] = (__bf16)v;
          }
        }
      }
    }
  }
}

// ---------------------------------------------------------------------------
// Flash attention (values == K, per the reference).  Q,K: [B*H, S, 64] bf16.
// Block = 256 thr (8 waves); each wave owns 16 query rows; key tiles of 32,
// double-buffered in LDS with async staging of the row-major copy.
// out: bf16 [NTOK, D] (column offset h*64).
// ---------------------------------------------------------------------------
__global__ void attention_wmma(const __bf16* __restrict__ Q, const __bf16* __restrict__ Kmat,
                               const int* __restrict__ mask, __bf16* __restrict__ outBF) {
  __shared__ __align__(16) __bf16 Krow[2][32][72];   // K tile row-major  [key][d]
  __shared__ __align__(16) __bf16 KtL[2][64][40];    // K tile transposed [d][key]
  __shared__ __align__(16) __bf16 Pst[8][16][32];    // per-wave softmax tile staging

  const int bh = blockIdx.y;
  const int b  = bh / Hh;
  const int h  = bh % Hh;
  const __bf16* Qb = Q    + (size_t)bh * Ss * DKk;
  const __bf16* Kb = Kmat + (size_t)bh * Ss * DKk;

  const int t    = threadIdx.x;
  const int wave = t >> 5;
  const int lane = t & 31;
  const int lr   = lane & 15;
  const int hi   = lane >> 4;
  const int lk   = hi * 16;
  const int q0   = blockIdx.x * 128 + wave * 16;

  // Q fragments for d-steps 0 and 32 (held in registers for the whole K loop)
  bf16x16 qf[2];
#pragma unroll
  for (int d2 = 0; d2 < 2; ++d2) {
    const __bf16* qr = Qb + (size_t)(q0 + lr) * DKk + d2 * 32;
#pragma unroll
    for (int j = 0; j < 8; ++j) { qf[d2][j] = qr[lk + j]; qf[d2][8 + j] = qr[lk + 8 + j]; }
  }

  float mrow[8], lsum[8];
  floatx8 o[4];
#pragma unroll
  for (int r = 0; r < 8; ++r) { mrow[r] = -3.0e38f; lsum[r] = 0.f; }
#pragma unroll
  for (int ns = 0; ns < 4; ++ns)
#pragma unroll
    for (int r = 0; r < 8; ++r) o[ns][r] = 0.f;

  // ---- K tile staging: async row-major copy + manual transposed copy ----
  const int kr  = t >> 3;        // 0..31 (key within tile)
  const int dsg = (t & 7) * 8;   // 0..56 (d segment)
  auto stageK = [&](int kk0, int buf) {
    const __bf16* src = Kb + (size_t)(kk0 + kr) * DKk + dsg;
    async_copy_b128(src, &Krow[buf][kr][dsg]);
#pragma unroll
    for (int j = 0; j < 8; ++j) KtL[buf][dsg + j][kr] = src[j];
  };

  const int NT = Ss / 32;
  stageK(0, 0);
  wait_async0();
  __syncthreads();

  for (int it = 0; it < NT; ++it) {
    const int kk0 = it * 32;
    const int cur = it & 1;
    if (it + 1 < NT) stageK(kk0 + 32, cur ^ 1);   // overlap next K tile with math

    // ---- scores: S = Q @ K^T  (two 16-key sub-tiles, K-dim 64 = 2 wmma) ----
    floatx8 sc[2];
#pragma unroll
    for (int sub = 0; sub < 2; ++sub) {
#pragma unroll
      for (int r = 0; r < 8; ++r) sc[sub][r] = 0.f;
#pragma unroll
      for (int d2 = 0; d2 < 2; ++d2) {
        bf16x16 bf;
        int key = sub * 16 + lr;
        int db  = d2 * 32 + hi * 16;
#pragma unroll
        for (int e = 0; e < 16; ++e) bf[e] = Krow[cur][key][db + e];
        sc[sub] = WMMA_BF16(qf[d2], bf, sc[sub]);
      }
    }

    // ---- scale + mask ----
    float mk[2];
#pragma unroll
    for (int sub = 0; sub < 2; ++sub) {
      int key = kk0 + sub * 16 + lr;
      mk[sub] = (mask[b * Ss + key] == 0) ? -1e9f : 0.f;
    }
    float tmax[8];
#pragma unroll
    for (int r = 0; r < 8; ++r) {
      float a0 = sc[0][r] * 0.125f + mk[0];
      float a1 = sc[1][r] * 0.125f + mk[1];
      sc[0][r] = a0; sc[1][r] = a1;
      tmax[r] = fmaxf(a0, a1);
    }
    // row-max across the 16-lane group (rows live per 16-lane half of wave32)
#pragma unroll
    for (int r = 0; r < 8; ++r) {
      float v = tmax[r];
#pragma unroll
      for (int off = 1; off < 16; off <<= 1) v = fmaxf(v, __shfl_xor(v, off, 16));
      tmax[r] = v;
    }

    // ---- online softmax update ----
    float rs[8];
#pragma unroll
    for (int r = 0; r < 8; ++r) {
      float newm  = fmaxf(mrow[r], tmax[r]);
      float alpha = __expf(mrow[r] - newm);
      mrow[r] = newm;
      lsum[r] *= alpha;
#pragma unroll
      for (int ns = 0; ns < 4; ++ns) o[ns][r] *= alpha;
      float p0 = __expf(sc[0][r] - newm);
      float p1 = __expf(sc[1][r] - newm);
      sc[0][r] = p0; sc[1][r] = p1;
      rs[r] = p0 + p1;
    }
#pragma unroll
    for (int r = 0; r < 8; ++r) {
      float v = rs[r];
#pragma unroll
      for (int off = 1; off < 16; off <<= 1) v += __shfl_xor(v, off, 16);
      lsum[r] += v;
    }

    // ---- restage P (C layout -> A layout) via per-wave LDS ----
#pragma unroll
    for (int sub = 0; sub < 2; ++sub)
#pragma unroll
      for (int r = 0; r < 8; ++r)
        Pst[wave][hi * 8 + r][sub * 16 + lr] = (__bf16)sc[sub][r];
    // Same-wave DS ops are in-order per the ISA; alias analysis keeps order.

    bf16x16 pf;
#pragma unroll
    for (int j = 0; j < 8; ++j) { pf[j] = Pst[wave][lr][lk + j]; pf[8 + j] = Pst[wave][lr][lk + 8 + j]; }

    // ---- O += P @ Ktile  (values == K) ----
#pragma unroll
    for (int ns = 0; ns < 4; ++ns) {
      bf16x16 bf;
      int col = ns * 16 + lr;
      int bk2 = hi * 16;
#pragma unroll
      for (int e = 0; e < 16; ++e) bf[e] = KtL[cur][col][bk2 + e];
      o[ns] = WMMA_BF16(pf, bf, o[ns]);
    }

    wait_async0();
    __syncthreads();
  }

  // ---- normalize + store ----
#pragma unroll
  for (int r = 0; r < 8; ++r) lsum[r] = 1.0f / lsum[r];
#pragma unroll
  for (int ns = 0; ns < 4; ++ns)
#pragma unroll
    for (int r = 0; r < 8; ++r) {
      int s_idx = q0 + hi * 8 + r;
      int col   = h * 64 + ns * 16 + lr;
      outBF[(size_t)(b * Ss + s_idx) * Dd + col] = (__bf16)(o[ns][r] * lsum[r]);
    }
}

// ---------------------------------------------------------------------------
// Host-side launch
// ---------------------------------------------------------------------------
extern "C" void kernel_launch(void* const* d_in, const int* in_sizes, int n_in,
                              void* d_out, int out_size, void* d_ws, size_t ws_size,
                              hipStream_t stream) {
  const float* x     = (const float*)d_in[0];
  const int*   mask  = (const int*)d_in[1];
  const float* Wq    = (const float*)d_in[2];
  const float* bq    = (const float*)d_in[3];
  const float* Wk    = (const float*)d_in[4];
  const float* bk    = (const float*)d_in[5];
  // d_in[6] = Wv, d_in[7] = bv : dead code in the reference (values == K)
  const float* Wo    = (const float*)d_in[8];
  const float* bo    = (const float*)d_in[9];
  const float* ln1_g = (const float*)d_in[10];
  const float* ln1_b = (const float*)d_in[11];
  const float* W1    = (const float*)d_in[12];
  const float* b1    = (const float*)d_in[13];
  const float* W2    = (const float*)d_in[14];
  const float* b2    = (const float*)d_in[15];
  const float* ln2_g = (const float*)d_in[16];
  const float* ln2_b = (const float*)d_in[17];
  float* out = (float*)d_out;

  char* ws = (char*)d_ws;
  size_t off = 0;
  auto alloc = [&](size_t bytes) { char* p = ws + off; off += (bytes + 255) & ~(size_t)255; return p; };

  __bf16* hidden1 = (__bf16*)alloc((size_t)NTOK * Dd * 2);
  __bf16* q_bf    = (__bf16*)alloc((size_t)NTOK * Dd * 2);
  __bf16* k_bf    = (__bf16*)alloc((size_t)NTOK * Dd * 2);
  __bf16* attn_bf = (__bf16*)alloc((size_t)NTOK * Dd * 2);
  float*  x2      = (float*)alloc((size_t)NTOK * Dd * 4);
  __bf16* hidden2 = (__bf16*)alloc((size_t)NTOK * Dd * 2);
  __bf16* ff1_bf  = (__bf16*)alloc((size_t)NTOK * DFFf * 2);
  __bf16* Wq_bf   = (__bf16*)alloc((size_t)Dd * Dd * 2);
  __bf16* Wk_bf   = (__bf16*)alloc((size_t)Dd * Dd * 2);
  __bf16* Wo_bf   = (__bf16*)alloc((size_t)Dd * Dd * 2);
  __bf16* W1_bf   = (__bf16*)alloc((size_t)Dd * DFFf * 2);
  __bf16* W2_bf   = (__bf16*)alloc((size_t)DFFf * Dd * 2);

  // weight conversion
  cvt_f32_bf16<<<1024, 256, 0, stream>>>(Wq, Wq_bf, Dd * Dd);
  cvt_f32_bf16<<<1024, 256, 0, stream>>>(Wk, Wk_bf, Dd * Dd);
  cvt_f32_bf16<<<1024, 256, 0, stream>>>(Wo, Wo_bf, Dd * Dd);
  cvt_f32_bf16<<<1024, 256, 0, stream>>>(W1, W1_bf, Dd * DFFf);
  cvt_f32_bf16<<<1024, 256, 0, stream>>>(W2, W2_bf, DFFf * Dd);

  // LN1
  layernorm_bf16<<<NTOK, 256, 0, stream>>>(x, ln1_g, ln1_b, hidden1);

  // Q / K projections -> [B,H,S,DK] bf16
  dim3 gProj(Dd / 128, NTOK / 128);
  gemm_bf16_wmma<<<gProj, 256, 0, stream>>>(hidden1, Wq_bf, bq, nullptr, nullptr, q_bf,
                                            NTOK, Dd, Dd, FLAG_HEAD);
  gemm_bf16_wmma<<<gProj, 256, 0, stream>>>(hidden1, Wk_bf, bk, nullptr, nullptr, k_bf,
                                            NTOK, Dd, Dd, FLAG_HEAD);

  // attention (values == K)
  attention_wmma<<<dim3(Ss / 128, Bb * Hh), 256, 0, stream>>>(q_bf, k_bf, mask, attn_bf);

  // output projection + residual -> x2 (fp32)
  gemm_bf16_wmma<<<gProj, 256, 0, stream>>>(attn_bf, Wo_bf, bo, x, x2, nullptr,
                                            NTOK, Dd, Dd, 0);

  // LN2
  layernorm_bf16<<<NTOK, 256, 0, stream>>>(x2, ln2_g, ln2_b, hidden2);

  // FFN1 (GELU) -> bf16
  dim3 gFf1(DFFf / 128, NTOK / 128);
  gemm_bf16_wmma<<<gFf1, 256, 0, stream>>>(hidden2, W1_bf, b1, nullptr, nullptr, ff1_bf,
                                           NTOK, DFFf, Dd, FLAG_GELU);

  // FFN2 + residual -> out (fp32)
  gemm_bf16_wmma<<<gProj, 256, 0, stream>>>(ff1_bf, W2_bf, b2, x2, out, nullptr,
                                            NTOK, Dd, DFFf, 0);
}
